// RecurrentBlock_85469849190426
// MI455X (gfx1250) — compile-verified, matrix operand order checked by reading, not run on hardware
//
#include <hip/hip_runtime.h>
#include <hip/hip_bf16.h>
#include <cstdint>
#include <cstddef>

#define B_   64
#define T_   1024
#define I_   256
#define H_   256
#define NG_  768
#define XOUT_ELEMS (B_*T_*H_)

typedef __attribute__((ext_vector_type(16))) __bf16 v16bf;
typedef __attribute__((ext_vector_type(8)))  float  v8f;
typedef __attribute__((ext_vector_type(8)))  int    v8i;

__device__ __forceinline__ unsigned short f2bf(float f) {
    union { float f; unsigned u; } x; x.f = f;
    unsigned r = x.u + 0x7fffu + ((x.u >> 16) & 1u);
    return (unsigned short)(r >> 16);
}
__device__ __forceinline__ float bf2f(unsigned short h) {
    union { float f; unsigned u; } x; x.u = ((unsigned)h) << 16; return x.f;
}
__device__ __forceinline__ unsigned char f2e4m3(float v) {
    union { float f; unsigned u; } x; x.f = v;
    unsigned s = (x.u >> 24) & 0x80u;
    float a = fminf(fabsf(v), 448.f);
    x.f = a;
    unsigned u = x.u + (1u << 19);
    int eb = (int)((u >> 23) & 0xffu) - 120;
    unsigned mant = (u >> 20) & 7u;
    unsigned code;
    if (eb <= 0) code = 0u;
    else { code = ((unsigned)eb << 3) | mant; if (code > 0x7eu) code = 0x7eu; }
    return (unsigned char)(s | code);
}
__device__ __forceinline__ float sigm_(float x) {
    x = fminf(fmaxf(x, -30.f), 30.f);
    return 1.f / (1.f + __expf(-x));
}
__device__ __forceinline__ float tanh_(float x) {
    x = fminf(fmaxf(x, -10.f), 10.f);
    float e = __expf(2.f * x);
    return (e - 1.f) / (e + 1.f);
}
__device__ __forceinline__ unsigned pk_bf16(float lo, float hi) {
    union { __hip_bfloat162 h; unsigned u; } p;
    p.h = __float22bfloat162_rn(make_float2(lo, hi));
    return p.u;
}

__global__ __launch_bounds__(256) void gi_gemm(const float* __restrict__ X,
                                               const float* __restrict__ Wih,
                                               const float* __restrict__ bih,
                                               unsigned short* __restrict__ gi)
{
    __shared__ __attribute__((aligned(32))) unsigned short sA[2][128][32];
    __shared__ __attribute__((aligned(32))) unsigned short sB[2][128][32];

    const int tid = threadIdx.x;
    const int bid = blockIdx.x;
    const int mt = bid / 6, nt = bid % 6;
    const int r0 = mt * 128, c0 = nt * 128;
    const int w  = tid >> 5, L = tid & 31, Lm = L & 15, Lh = L >> 4;
    const int wm = w >> 2,  wn = w & 3;

    const v8f vzero = {0.f,0.f,0.f,0.f,0.f,0.f,0.f,0.f};
    v8f acc[4][2];
    #pragma unroll
    for (int i = 0; i < 4; i++)
        #pragma unroll
        for (int j = 0; j < 2; j++) acc[i][j] = vzero;

    auto load_stage = [&](int ks, int buf) {
        const int k0 = ks * 32;
        float4 ra[4], rb[4];
        #pragma unroll
        for (int i = 0; i < 4; i++) {
            int q = tid + i * 256;
            int row = q >> 3, seg = q & 7;
            ra[i] = *(const float4*)&X  [(size_t)(r0 + row) * I_ + k0 + seg * 4];
            rb[i] = *(const float4*)&Wih[(size_t)(c0 + row) * I_ + k0 + seg * 4];
        }
        #pragma unroll
        for (int i = 0; i < 4; i++) {
            int q = tid + i * 256;
            int row = q >> 3, seg = q & 7;
            *(uint2*)&sA[buf][row][seg * 4] =
                make_uint2(pk_bf16(ra[i].x, ra[i].y), pk_bf16(ra[i].z, ra[i].w));
            *(uint2*)&sB[buf][row][seg * 4] =
                make_uint2(pk_bf16(rb[i].x, rb[i].y), pk_bf16(rb[i].z, rb[i].w));
        }
    };
    auto compute = [&](int buf) {
        v16bf a[4], bb[2];
        #pragma unroll
        for (int mf = 0; mf < 4; mf++)
            a[mf] = *(const v16bf*)&sA[buf][wm * 64 + mf * 16 + Lm][Lh * 16];
        #pragma unroll
        for (int nf = 0; nf < 2; nf++)
            bb[nf] = *(const v16bf*)&sB[buf][wn * 32 + nf * 16 + Lm][Lh * 16];
        #pragma unroll
        for (int mf = 0; mf < 4; mf++)
            #pragma unroll
            for (int nf = 0; nf < 2; nf++)
                acc[mf][nf] = __builtin_amdgcn_wmma_f32_16x16x32_bf16(
                    false, a[mf], false, bb[nf], (short)0, acc[mf][nf], false, false);
    };

    load_stage(0, 0);
    __syncthreads();
    for (int ks = 0; ks < 8; ks++) {
        if (ks < 7) load_stage(ks + 1, (ks + 1) & 1);
        compute(ks & 1);
        __syncthreads();
    }

    #pragma unroll
    for (int nf = 0; nf < 2; nf++) {
        const int col = c0 + wn * 32 + nf * 16 + Lm;
        const float bias = bih[col];
        #pragma unroll
        for (int mf = 0; mf < 4; mf++) {
            #pragma unroll
            for (int v = 0; v < 8; v++) {
                int r = r0 + wm * 64 + mf * 16 + Lh * 8 + v;
                int b = r >> 10, t = r & (T_ - 1);
                gi[(size_t)(t * B_ + b) * NG_ + col] = f2bf(acc[mf][nf][v] + bias);
            }
        }
    }
}

__global__ __launch_bounds__(512) void gru_scan(const unsigned short* __restrict__ gi,
                                                const float* __restrict__ st,
                                                const float* __restrict__ msk,
                                                const float* __restrict__ Whh,
                                                const float* __restrict__ bhh,
                                                float* __restrict__ out)
{
    __shared__ __attribute__((aligned(32))) unsigned char  sWrz[32][4][32][32];
    __shared__ __attribute__((aligned(32))) unsigned short sWn [16][8][32][16];
    __shared__ __attribute__((aligned(32))) unsigned short sHbf[16][256];
    __shared__ __attribute__((aligned(32))) unsigned char  sHf8[16][256];
    __shared__ __attribute__((aligned(32))) float          sOut[16][256];
    __shared__ __attribute__((aligned(16))) unsigned short sGi[16 * NG_];
    __shared__ float sMsk[16];

    const int tid = threadIdx.x;
    const int bg  = blockIdx.x;
    const int w   = tid >> 5, L = tid & 31, Lm = L & 15, Lh = L >> 4;
    const v8f vzero = {0.f,0.f,0.f,0.f,0.f,0.f,0.f,0.f};

    for (int idx = tid; idx < 16 * 8 * 32 * 16; idx += 512) {
        int e = idx & 15, L2 = (idx >> 4) & 31, kf = (idx >> 9) & 7, ww = idx >> 12;
        int col = 512 + ww * 16 + (L2 & 15);
        int k   = kf * 32 + (L2 >> 4) * 16 + e;
        sWn[ww][kf][L2][e] = f2bf(Whh[(size_t)col * H_ + k]);
    }
    for (int idx = tid; idx < 32 * 4 * 32 * 32; idx += 512) {
        int j = idx & 31, L2 = (idx >> 5) & 31, kf = (idx >> 10) & 3, p = idx >> 12;
        int g = p >> 4, wc = p & 15;
        int col = g * 256 + wc * 16 + (L2 & 15);
        int k   = kf * 64 + (L2 >> 4) * 32 + j;
        sWrz[p][kf][L2][j] = f2e4m3(Whh[(size_t)col * H_ + k] * 256.f);
    }
    for (int idx = tid; idx < 4096; idx += 512) {
        int m = idx >> 8, k = idx & 255, b = bg * 16 + m;
        float h0 = st[(size_t)b * H_ + k] * msk[(size_t)b * T_];
        sHbf[m][k] = f2bf(h0);
        sHf8[m][k] = f2e4m3(h0 * 16.f);
    }
    {
        const char* slab = (const char*)gi + (size_t)(bg * 16) * NG_ * 2;
        #pragma unroll
        for (int i = 0; i < 3; i++)
            *(float4*)((char*)sGi + tid * 48 + i * 16) =
                *(const float4*)(slab + tid * 48 + i * 16);
        if (tid < 16) sMsk[tid] = msk[(size_t)(bg * 16 + tid) * T_];
    }
    const int c_ = w * 16 + Lm;
    const float br = bhh[c_], bz = bhh[256 + c_], bn = bhh[512 + c_];
    float hP[8];
    #pragma unroll
    for (int v = 0; v < 8; v++) {
        int b = bg * 16 + Lh * 8 + v;
        hP[v] = st[(size_t)b * H_ + c_] * msk[(size_t)b * T_];
    }
    __syncthreads();

    for (int t = 0; t < T_; t++) {
        const bool have = (t + 1 < T_);
        float4 giReg[3];
        float  mReg = 0.f;
        if (have) {
            const char* slab = (const char*)gi + (size_t)((t + 1) * B_ + bg * 16) * NG_ * 2;
            #pragma unroll
            for (int i = 0; i < 3; i++)
                giReg[i] = *(const float4*)(slab + tid * 48 + i * 16);
            if (tid < 16) mReg = msk[(size_t)(bg * 16 + tid) * T_ + t + 1];
        }

        v16bf aBF[8];
        #pragma unroll
        for (int kf = 0; kf < 8; kf++)
            aBF[kf] = *(const v16bf*)&sHbf[Lm][kf * 32 + Lh * 16];
        v8i aF8[4];
        #pragma unroll
        for (int kf = 0; kf < 4; kf++)
            aF8[kf] = *(const v8i*)&sHf8[Lm][kf * 64 + Lh * 32];

        v8f rA = vzero, zA = vzero, nA = vzero;
        #pragma unroll
        for (int kf = 0; kf < 4; kf++) {
            rA = __builtin_amdgcn_wmma_f32_16x16x64_fp8_fp8(
                     aF8[kf], *(const v8i*)&sWrz[w][kf][L][0],      (short)0, rA, false, false);
            zA = __builtin_amdgcn_wmma_f32_16x16x64_fp8_fp8(
                     aF8[kf], *(const v8i*)&sWrz[16 + w][kf][L][0], (short)0, zA, false, false);
        }
        #pragma unroll
        for (int kf = 0; kf < 8; kf++)
            nA = __builtin_amdgcn_wmma_f32_16x16x32_bf16(
                     false, aBF[kf], false, *(const v16bf*)&sWn[w][kf][L][0],
                     (short)0, nA, false, false);

        float nhv[8];
        #pragma unroll
        for (int v = 0; v < 8; v++) {
            int m = Lh * 8 + v;
            float gir = bf2f(sGi[m * NG_ + c_]);
            float giz = bf2f(sGi[m * NG_ + 256 + c_]);
            float gin = bf2f(sGi[m * NG_ + 512 + c_]);
            float mt  = sMsk[m];
            float r = sigm_(gir + rA[v] * (1.f / 4096.f) + br);
            float z = sigm_(giz + zA[v] * (1.f / 4096.f) + bz);
            float n = tanh_(gin + r * (nA[v] + bn));
            float nh = (1.f - z) * n + z * hP[v];
            float hn = hP[v] + (nh - hP[v]) * mt;
            hP[v] = hn;
            nhv[v] = nh;
        }
        __syncthreads();
        #pragma unroll
        for (int v = 0; v < 8; v++) {
            int m = Lh * 8 + v;
            sHbf[m][c_] = f2bf(hP[v]);
            sHf8[m][c_] = f2e4m3(hP[v] * 16.f);
            sOut[m][c_] = nhv[v];
        }
        if (have) {
            #pragma unroll
            for (int i = 0; i < 3; i++)
                *(float4*)((char*)sGi + tid * 48 + i * 16) = giReg[i];
            if (tid < 16) sMsk[tid] = mReg;
        }
        __syncthreads();
        for (int idx = tid; idx < 4096; idx += 512) {
            int m = idx >> 8, c = idx & 255;
            out[((size_t)(bg * 16 + m) * T_ + t) * H_ + c] = sOut[m][c];
        }
    }

    __syncthreads();
    #pragma unroll
    for (int v = 0; v < 8; v++) sOut[Lh * 8 + v][c_] = hP[v];
    __syncthreads();
    for (int idx = tid; idx < 4096; idx += 512) {
        int m = idx >> 8, c = idx & 255;
        out[(size_t)XOUT_ELEMS + (size_t)(bg * 16 + m) * H_ + c] = sOut[m][c];
    }
}

extern "C" void kernel_launch(void* const* d_in, const int* in_sizes, int n_in,
                              void* d_out, int out_size, void* d_ws, size_t ws_size,
                              hipStream_t stream)
{
    (void)in_sizes; (void)n_in; (void)out_size; (void)ws_size;
    const float* X   = (const float*)d_in[0];
    const float* st  = (const float*)d_in[1];
    const float* msk = (const float*)d_in[2];
    const float* Wih = (const float*)d_in[3];
    const float* Whh = (const float*)d_in[4];
    const float* bih = (const float*)d_in[5];
    const float* bhh = (const float*)d_in[6];
    float* out = (float*)d_out;
    unsigned short* gi = (unsigned short*)d_ws;

    hipLaunchKernelGGL(gi_gemm, dim3((B_ * T_ / 128) * (NG_ / 128)), dim3(256), 0, stream,
                       X, Wih, bih, gi);
    hipLaunchKernelGGL(gru_scan, dim3(B_ / 16), dim3(512), 0, stream,
                       gi, st, msk, Whh, bhh, out);
}
